// DecoderRNN_34694745817437
// MI455X (gfx1250) — compile-verified, matrix operand order checked by reading, not run on hardware
//
#include <hip/hip_runtime.h>
#include <hip/hip_bf16.h>

typedef __bf16 bf16;
typedef __attribute__((ext_vector_type(16))) __bf16 v16bf;
typedef __attribute__((ext_vector_type(8)))  float  v8f;

// B=32, T=64 (63 steps), S=64, V=32000, E=512, H=1024
#define NB   32
#define NT   63
#define NS   64
#define NV   32000
#define NE   512
#define NH   1024
#define NG   4096   // 4*H

// ---------------------------------------------------------------------------
// f32 -> bf16 bulk convert (grid-stride)
// ---------------------------------------------------------------------------
__global__ void cvt_f32_bf16(const float* __restrict__ src, bf16* __restrict__ dst, int n) {
    int i = blockIdx.x * blockDim.x + threadIdx.x;
    int stride = gridDim.x * blockDim.x;
    for (; i < n; i += stride) dst[i] = (bf16)src[i];
}

__global__ void bias_sum(const float* __restrict__ a, const float* __restrict__ b,
                         float* __restrict__ o) {
    int i = blockIdx.x * blockDim.x + threadIdx.x;
    if (i < NG) o[i] = a[i] + b[i];
}

// gather embedding rows for all 63 steps into bf16 x buffer: row = t*32 + b
__global__ void gather_emb(const int* __restrict__ tgt, const float* __restrict__ emb,
                           bf16* __restrict__ x) {
    int idx = blockIdx.x * blockDim.x + threadIdx.x;
    if (idx >= NT * NB * NE) return;
    int col = idx & (NE - 1);
    int row = idx >> 9;           // /512
    int t   = row >> 5;           // /32
    int b   = row & 31;
    int tok = tgt[b * 64 + t];    // tgt_input[b][t], T=64, use [:, :-1]
    x[idx] = (bf16)emb[(size_t)tok * NE + col];
}

__global__ void init_state(const float* __restrict__ hs, const float* __restrict__ cs,
                           float* __restrict__ h, float* __restrict__ c,
                           bf16* __restrict__ h_bf) {
    int i = blockIdx.x * blockDim.x + threadIdx.x;
    if (i < NB * NH) { h[i] = hs[i]; c[i] = cs[i]; h_bf[i] = (bf16)hs[i]; }
}

// ---------------------------------------------------------------------------
// WMMA GEMM: C[m][n] = sum_k A[m][k]*B[n][k] (+bias[n]) (+Cin[m][n]) (tanh)
// A: M x K bf16 row-major, B: N x K bf16 row-major.
// One wave = one 32x16 output tile: two f32 accumulators share one B fragment
// per k-step (halves B traffic; 3 b128 loads per WMMA instead of 4).
// Grid must exactly cover (M/32)*(N/16) waves so EXEC stays all-ones.
// ---------------------------------------------------------------------------
__global__ void __launch_bounds__(128)
gemm_bt_wmma(const bf16* __restrict__ A, int lda,
             const bf16* __restrict__ B, int ldb,
             const float* __restrict__ bias,
             const float* __restrict__ Cin, int ldcin,
             float* __restrict__ Cf, long long ldcf,
             bf16* __restrict__ Cbf, int ldcbf,
             int N, int K, int act) {
    const int lane  = threadIdx.x & 31;
    const int wave  = (blockIdx.x * blockDim.x + threadIdx.x) >> 5;
    const int nTiles = N >> 4;
    const int mTile = wave / nTiles;        // 32-row super-tile index
    const int nTile = wave - mTile * nTiles;
    const int half  = lane >> 4;   // 0: K 0-7 & 16-23 ; 1: K 8-15 & 24-31
    const int lr    = lane & 15;

    const bf16* pA0 = A + (size_t)(mTile * 32 + lr)      * lda + half * 8;
    const bf16* pA1 = A + (size_t)(mTile * 32 + 16 + lr) * lda + half * 8;
    const bf16* pB  = B + (size_t)(nTile * 16 + lr)      * ldb + half * 8;

    v8f acc0 = {}, acc1 = {};
    for (int k = 0; k < K; k += 32) {
        v16bf a0, a1, bv;
        uint4* a0p = reinterpret_cast<uint4*>(&a0);
        uint4* a1p = reinterpret_cast<uint4*>(&a1);
        uint4* bp  = reinterpret_cast<uint4*>(&bv);
        a0p[0] = *reinterpret_cast<const uint4*>(pA0 + k);
        a0p[1] = *reinterpret_cast<const uint4*>(pA0 + k + 16);
        a1p[0] = *reinterpret_cast<const uint4*>(pA1 + k);
        a1p[1] = *reinterpret_cast<const uint4*>(pA1 + k + 16);
        bp[0]  = *reinterpret_cast<const uint4*>(pB + k);
        bp[1]  = *reinterpret_cast<const uint4*>(pB + k + 16);
        acc0 = __builtin_amdgcn_wmma_f32_16x16x32_bf16(
                   false, a0, false, bv, (short)0, acc0, false, false);
        acc1 = __builtin_amdgcn_wmma_f32_16x16x32_bf16(
                   false, a1, false, bv, (short)0, acc1, false, false);
    }

    // C/D layout: VGPR v -> M = base + half*8 + v ; N = nTile*16 + (lane&15)
    const int n      = nTile * 16 + lr;
    const int mBase0 = mTile * 32 + half * 8;
    const int mBase1 = mBase0 + 16;
    const float bval = bias ? bias[n] : 0.0f;
#pragma unroll
    for (int v = 0; v < 8; ++v) {
        int m0 = mBase0 + v, m1 = mBase1 + v;
        float r0 = acc0[v] + bval;
        float r1 = acc1[v] + bval;
        if (Cin) { r0 += Cin[(size_t)m0 * ldcin + n];
                   r1 += Cin[(size_t)m1 * ldcin + n]; }
        if (act) { r0 = tanhf(r0); r1 = tanhf(r1); }
        if (Cf)  { Cf [(size_t)m0 * ldcf  + n] = r0;
                   Cf [(size_t)m1 * ldcf  + n] = r1; }
        if (Cbf) { Cbf[(size_t)m0 * ldcbf + n] = (bf16)r0;
                   Cbf[(size_t)m1 * ldcbf + n] = (bf16)r1; }
    }
}

// ---------------------------------------------------------------------------
// LSTM cell: gates (32 x 4096) -> update c, h ; emit h fp32, h bf16, hc_cat lo
// ---------------------------------------------------------------------------
__global__ void lstm_cell(const float* __restrict__ gates, float* __restrict__ c,
                          float* __restrict__ h, bf16* __restrict__ h_bf,
                          bf16* __restrict__ hc_cat) {
    int idx = blockIdx.x * blockDim.x + threadIdx.x;   // 32*1024
    int b = idx >> 10, j = idx & (NH - 1);
    const float* g = gates + (size_t)b * NG;
    float gi = g[j], gf = g[NH + j], gg = g[2 * NH + j], go = g[3 * NH + j];
    float si = 1.f / (1.f + __expf(-gi));
    float sf = 1.f / (1.f + __expf(-gf));
    float so = 1.f / (1.f + __expf(-go));
    float cn = sf * c[idx] + si * tanhf(gg);
    float hn = so * tanhf(cn);
    c[idx] = cn;
    h[idx] = hn;
    h_bf[idx] = (bf16)hn;
    hc_cat[(size_t)b * (2 * NH) + j] = (bf16)hn;
}

// ---------------------------------------------------------------------------
// Attention: scores = h . enc[b,s,:], softmax over S=64, context -> hc_cat hi
// One block per batch element; 8 waves; shfl reductions.
// ---------------------------------------------------------------------------
__global__ void __launch_bounds__(256)
attn_kernel(const float* __restrict__ h, const bf16* __restrict__ enc,
            bf16* __restrict__ hc_cat) {
    __shared__ float sh_h[NH];
    __shared__ float sh_sc[NS];
    __shared__ float sh_at[NS];
    const int b = blockIdx.x;
    const int t = threadIdx.x;
#pragma unroll
    for (int i = 0; i < 4; ++i) sh_h[t + i * 256] = h[(size_t)b * NH + t + i * 256];
    __syncthreads();

    const int wid = t >> 5, lane = t & 31;
    for (int it = 0; it < 8; ++it) {
        int s = wid * 8 + it;
        const bf16* e = enc + ((size_t)b * NS + s) * NH;
        float p = 0.f;
        for (int k = lane; k < NH; k += 32) p += sh_h[k] * (float)e[k];
        for (int off = 16; off > 0; off >>= 1) p += __shfl_xor(p, off, 32);
        if (lane == 0) sh_sc[s] = p;
    }
    __syncthreads();

    if (t < 32) {
        float s0 = sh_sc[t], s1 = sh_sc[t + 32];
        float mx = fmaxf(s0, s1);
        for (int off = 16; off > 0; off >>= 1) mx = fmaxf(mx, __shfl_xor(mx, off, 32));
        float e0 = __expf(s0 - mx), e1 = __expf(s1 - mx);
        float sm = e0 + e1;
        for (int off = 16; off > 0; off >>= 1) sm += __shfl_xor(sm, off, 32);
        float inv = 1.f / sm;
        sh_at[t] = e0 * inv;
        sh_at[t + 32] = e1 * inv;
    }
    __syncthreads();

#pragma unroll
    for (int i = 0; i < 4; ++i) {
        int hh = t + i * 256;
        float ctx = 0.f;
        for (int s = 0; s < NS; ++s)
            ctx += sh_at[s] * (float)enc[((size_t)b * NS + s) * NH + hh];
        hc_cat[(size_t)b * (2 * NH) + NH + hh] = (bf16)ctx;
    }
}

// ---------------------------------------------------------------------------
// Host driver
// ---------------------------------------------------------------------------
extern "C" void kernel_launch(void* const* d_in, const int* in_sizes, int n_in,
                              void* d_out, int out_size, void* d_ws, size_t ws_size,
                              hipStream_t stream) {
    const int*   tgt   = (const int*)  d_in[0];
    const float* hs    = (const float*)d_in[1];
    const float* cs    = (const float*)d_in[2];
    const float* enc   = (const float*)d_in[3];
    const float* emb   = (const float*)d_in[4];
    const float* W_ih  = (const float*)d_in[5];
    const float* W_hh  = (const float*)d_in[6];
    const float* b_ih  = (const float*)d_in[7];
    const float* b_hh  = (const float*)d_in[8];
    const float* W_w   = (const float*)d_in[9];
    const float* b_w   = (const float*)d_in[10];
    const float* W_out = (const float*)d_in[11];
    const float* b_out = (const float*)d_in[12];
    float* out = (float*)d_out;

    // workspace carve (~123 MB total; all bf16 weights ~82 MB -> L2-resident)
    char* ws = (char*)d_ws;
    size_t off = 0;
    auto carve = [&](size_t bytes) -> void* {
        off = (off + 255) & ~(size_t)255;
        void* p = ws + off;
        off += bytes;
        return p;
    };
    bf16*  W_ih_bf  = (bf16*) carve((size_t)NG * NE * 2);        //  4 MB
    bf16*  W_hh_bf  = (bf16*) carve((size_t)NG * NH * 2);        //  8 MB
    bf16*  W_w_bf   = (bf16*) carve((size_t)NH * 2 * NH * 2);    //  4 MB
    bf16*  W_out_bf = (bf16*) carve((size_t)NV * NH * 2);        // 62.5 MB
    bf16*  enc_bf   = (bf16*) carve((size_t)NB * NS * NH * 2);   //  4 MB
    bf16*  x_bf     = (bf16*) carve((size_t)NT * NB * NE * 2);   //  2 MB
    float* x_part   = (float*)carve((size_t)NT * NB * NG * 4);   // 31.5 MB
    float* bsum     = (float*)carve((size_t)NG * 4);
    float* h        = (float*)carve((size_t)NB * NH * 4);
    float* c        = (float*)carve((size_t)NB * NH * 4);
    bf16*  h_bf     = (bf16*) carve((size_t)NB * NH * 2);
    float* gates    = (float*)carve((size_t)NB * NG * 4);
    bf16*  hc_cat   = (bf16*) carve((size_t)NB * 2 * NH * 2);
    bf16*  dec_bf   = (bf16*) carve((size_t)NB * NH * 2);

    auto cvt = [&](const float* s, bf16* d, int n) {
        cvt_f32_bf16<<<(n + 255) / 256, 256, 0, stream>>>(s, d, n);
    };
    // one-time per-call prep
    cvt(W_ih,  W_ih_bf,  NG * NE);
    cvt(W_hh,  W_hh_bf,  NG * NH);
    cvt(W_w,   W_w_bf,   NH * 2 * NH);
    cvt(W_out, W_out_bf, NV * NH);
    cvt(enc,   enc_bf,   NB * NS * NH);
    bias_sum<<<NG / 256, 256, 0, stream>>>(b_ih, b_hh, bsum);
    gather_emb<<<(NT * NB * NE + 255) / 256, 256, 0, stream>>>(tgt, emb, x_bf);
    init_state<<<NB * NH / 256, 256, 0, stream>>>(hs, cs, h, c, h_bf);

    auto gemm = [&](const bf16* A, int lda, const bf16* B, int ldb,
                    const float* bias, const float* Cin, int ldcin,
                    float* Cf, long long ldcf, bf16* Cbf, int ldcbf,
                    int M, int N, int K, int act) {
        int waves = (M / 32) * (N / 16);        // exact cover, 4 waves/block
        gemm_bt_wmma<<<waves / 4, 128, 0, stream>>>(
            A, lda, B, ldb, bias, Cin, ldcin, Cf, ldcf, Cbf, ldcbf, N, K, act);
    };

    // hoisted input projection for all timesteps: x_part = x @ W_ih^T + (b_ih+b_hh)
    gemm(x_bf, NE, W_ih_bf, NE, bsum, nullptr, 0,
         x_part, NG, nullptr, 0, NT * NB, NG, NE, 0);

    // sequential decode
    for (int t = 0; t < NT; ++t) {
        // gates = h @ W_hh^T + x_part[t]
        gemm(h_bf, NH, W_hh_bf, NH, nullptr, x_part + (size_t)t * NB * NG, NG,
             gates, NG, nullptr, 0, NB, NG, NH, 0);
        lstm_cell<<<NB * NH / 256, 256, 0, stream>>>(gates, c, h, h_bf, hc_cat);
        attn_kernel<<<NB, 256, 0, stream>>>(h, enc_bf, hc_cat);
        // dec = tanh([h;ctx] @ W_w^T + b_w) -> bf16
        gemm(hc_cat, 2 * NH, W_w_bf, 2 * NH, b_w, nullptr, 0,
             nullptr, 0, dec_bf, NH, NB, NH, 2 * NH, 1);
        // logits = dec @ W_out^T + b_out -> out[b][t][v], row stride 63*32000
        gemm(dec_bf, NH, W_out_bf, NH, b_out, nullptr, 0,
             out + (size_t)t * NV, (long long)NT * NV, nullptr, 0,
             NB, NV, NH, 0);
    }
    (void)in_sizes; (void)n_in; (void)out_size; (void)ws_size;
}
